// CausalMultiheadAttention_50483045597371
// MI455X (gfx1250) — matrix operand order repaired and checked by reading, not compiled, not run on hardware
//
#include <hip/hip_runtime.h>
#include <hip/hip_bf16.h>

// ---------------------------------------------------------------------------
// CDNA5 (gfx1250) causal MHA: bf16 WMMA + Tensor Data Mover (TDM) tile loads.
// B=4, S=2048, D=1024, H=16, DK=64.
// ---------------------------------------------------------------------------

typedef __attribute__((ext_vector_type(16))) __bf16 v16bf;
typedef __attribute__((ext_vector_type(8)))  float  v8f;
typedef __attribute__((ext_vector_type(4)))  unsigned int v4u;
typedef __attribute__((ext_vector_type(8)))  int v8i;
typedef __attribute__((ext_vector_type(4)))  int v4i;

union Frag16 { uint4 q[2]; v16bf v; };

__device__ __forceinline__ v8f wmma_bf16(v16bf a, v16bf b, v8f c) {
  return __builtin_amdgcn_wmma_f32_16x16x32_bf16(false, a, false, b, (short)0, c,
                                                 false, false);
}

// Low 32 bits of a generic pointer to a __shared__ object = LDS byte address.
__device__ __forceinline__ unsigned lds_off(const void* p) {
  return (unsigned)(unsigned long long)p;
}

// ---------------------------------------------------------------------------
// TDM 2D tile load: global (row stride stride_elems, bf16) -> LDS.
// Optional LDS row padding: pad_interval_code k pads after 2^(k+1) dwords,
// pad_amount_code a adds (a+1) dwords. Descriptor per CDNA5 ISA 8.3-8.6.
// Issue from ONE wave (uniform branch); EXEC is ignored by TDM.
// Toolchain here is the 6-arg builtin: (g0 u32x4, g1 i32x8, i32x4, i32x4,
// i32x8, cpol) -- trailing groups zero for a 2D, count=1 descriptor.
// ---------------------------------------------------------------------------
__device__ __forceinline__ void tdm_load_2d(unsigned lds_addr, const void* gptr,
                                            unsigned tile_w, unsigned tile_h,
                                            unsigned stride_elems,
                                            bool pad_en, unsigned pad_interval_code,
                                            unsigned pad_amount_code) {
  unsigned long long ga = (unsigned long long)(size_t)gptr;
  v4u g0;
  g0[0] = 1u;                                          // count=1, user mode
  g0[1] = lds_addr;                                    // lds_addr [63:32]
  g0[2] = (unsigned)(ga & 0xffffffffull);              // global_addr[31:0]
  g0[3] = (unsigned)((ga >> 32) & 0x1ffffffull)        // global_addr[56:32]
          | (2u << 30);                                // type=2 ("image")
  unsigned d0 = (1u << 16);                            // data_size = 2 bytes
  if (pad_en)
    d0 |= (1u << 20) | (pad_interval_code << 22) | (pad_amount_code << 25);
  v8i g1;
  g1[0] = (int)d0;
  g1[1] = (int)((tile_w & 0xffffu) << 16);             // tensor_dim0[15:0] @63:48
  g1[2] = (int)((tile_w >> 16) & 0xffffu)              // tensor_dim0[31:16]
          | (int)((tile_h & 0xffffu) << 16);           // tensor_dim1[15:0]
  g1[3] = (int)((tile_h >> 16) & 0xffffu)              // tensor_dim1[31:16]
          | (int)(tile_w << 16);                       // tile_dim0 @127:112
  g1[4] = (int)(tile_h & 0xffffu);                     // tile_dim1; tile_dim2=0
  g1[5] = (int)stride_elems;                           // tensor_dim0_stride[31:0]
  g1[6] = 0;                                           // stride hi / dim1_stride lo
  g1[7] = 0;
  const v4i z4 = {0, 0, 0, 0};
  const v8i z8 = {0, 0, 0, 0, 0, 0, 0, 0};
  __builtin_amdgcn_tensor_load_to_lds(g0, g1, z4, z4, z8, 0);
}

// ---------------------------------------------------------------------------
// fp32 -> bf16 cast
// ---------------------------------------------------------------------------
__global__ void cast_f32_bf16(const float* __restrict__ in,
                              __bf16* __restrict__ out, int n) {
  int i = blockIdx.x * blockDim.x + threadIdx.x;
  if (i < n) out[i] = (__bf16)in[i];
}

// ---------------------------------------------------------------------------
// C[M,N] = A[M,K] * B[N,K]^T   (A,B bf16 row-major; C fp32 or bf16)
// Block: 128 threads = 4 waves; tile 64(M) x 128(N) x 32(K), TDM double-buffered.
// Wave (wm,wn) owns a 32x64 patch = 2x4 WMMA accumulators (8 WMMA / K-step).
// ---------------------------------------------------------------------------
template <bool OUT_BF16>
__global__ __launch_bounds__(128) void gemm_bf16_kernel(
    const __bf16* __restrict__ A, const __bf16* __restrict__ Bm,
    void* __restrict__ Cv, int M, int N, int K) {
  __shared__ __bf16 As[2][64 * 32];
  __shared__ __bf16 Bs[2][128 * 32];

  const int tid  = threadIdx.x;
  const int wave = tid >> 5;
  const int lane = tid & 31;
  const int ln = lane & 15, kg = lane >> 4, kb = kg * 8;
  const int m0 = blockIdx.y * 64, n0 = blockIdx.x * 128;
  const int wm = (wave >> 1) * 32, wn = (wave & 1) * 64;
  const bool issuer = (wave == 0);

  const v8f vzero = {0.f, 0.f, 0.f, 0.f, 0.f, 0.f, 0.f, 0.f};
  v8f acc[2][4];
#pragma unroll
  for (int i = 0; i < 2; ++i)
#pragma unroll
    for (int j = 0; j < 4; ++j) acc[i][j] = vzero;

  const int nk = K / 32;
  if (issuer) {  // prologue: fill buffer 0
    tdm_load_2d(lds_off(As[0]), A + (size_t)m0 * K, 32, 64, K, false, 0, 0);
    tdm_load_2d(lds_off(Bs[0]), Bm + (size_t)n0 * K, 32, 128, K, false, 0, 0);
  }

  for (int i = 0; i < nk; ++i) {
    const int cur = i & 1;
    if (issuer) {
      if (i + 1 < nk) {  // prefetch next tile, then wait for current (<=2 inflight)
        tdm_load_2d(lds_off(As[cur ^ 1]), A + (size_t)m0 * K + (i + 1) * 32,
                    32, 64, K, false, 0, 0);
        tdm_load_2d(lds_off(Bs[cur ^ 1]), Bm + (size_t)n0 * K + (i + 1) * 32,
                    32, 128, K, false, 0, 0);
        __builtin_amdgcn_s_wait_tensorcnt(2);
      } else {
        __builtin_amdgcn_s_wait_tensorcnt(0);
      }
    }
    __syncthreads();  // current buffer ready for all waves

    Frag16 af[2], bfg[4];
#pragma unroll
    for (int ii = 0; ii < 2; ++ii) {  // A 16x32: lane=row, K {kb..kb+7, kb+16..kb+23}
      const __bf16* p = As[cur] + (wm + ii * 16 + ln) * 32 + kb;
      af[ii].q[0] = *(const uint4*)(p);
      af[ii].q[1] = *(const uint4*)(p + 16);
    }
#pragma unroll
    for (int j = 0; j < 4; ++j) {     // B^T rows: lane=col n, K {kb..kb+7, kb+16..kb+23}
      const __bf16* p = Bs[cur] + (wn + j * 16 + ln) * 32 + kb;
      bfg[j].q[0] = *(const uint4*)(p);
      bfg[j].q[1] = *(const uint4*)(p + 16);
    }
#pragma unroll
    for (int ii = 0; ii < 2; ++ii)
#pragma unroll
      for (int j = 0; j < 4; ++j)
        acc[ii][j] = wmma_bf16(af[ii].v, bfg[j].v, acc[ii][j]);

    __syncthreads();  // all reads done before this buffer is refilled
  }

  // C layout: VGPR r -> row 8*kg + r, lane ln -> col
#pragma unroll
  for (int ii = 0; ii < 2; ++ii)
#pragma unroll
    for (int j = 0; j < 4; ++j)
#pragma unroll
      for (int r = 0; r < 8; ++r) {
        const int m = m0 + wm + ii * 16 + 8 * kg + r;
        const int n = n0 + wn + j * 16 + ln;
        if (OUT_BF16)
          ((__bf16*)Cv)[(size_t)m * N + n] = (__bf16)acc[ii][j][r];
        else
          ((float*)Cv)[(size_t)m * N + n] = acc[ii][j][r];
      }
}

// ---------------------------------------------------------------------------
// Flash attention with causal mask. qkv layout: [B*S, 3*H*DK] bf16
// (col = qsel*1024 + h*64 + dk). Output attn: [B*S, H*DK] bf16.
// One block per (qtile=64, h, b); 128 threads = 4 waves; wave owns 16 q rows.
// K tile loaded by TDM with hardware LDS padding (128B rows -> 144B stride).
// ---------------------------------------------------------------------------
__global__ __launch_bounds__(128) void flash_attn_kernel(
    const __bf16* __restrict__ qkv, __bf16* __restrict__ attn) {
  __shared__ __bf16 Ks[64 * 72];       // [key][dk], TDM-padded rows
  __shared__ __bf16 Vt[64 * 72];       // [dk][key]  (transposed)
  __shared__ __bf16 Pw[4][16 * 72];    // per-wave P tile [qrow][key]

  const int b = blockIdx.z, h = blockIdx.y, qt = blockIdx.x;
  const int tid  = threadIdx.x;
  const int wave = tid >> 5;
  const int lane = tid & 31;
  const int ln = lane & 15;
  const int kg = lane >> 4;
  const int kb = kg * 8;
  const int q_base = qt * 64;
  const float NEG_INF = -__builtin_inff();

  // Q fragments: 16x64, two 16x32 k-chunks (A layout), straight from global
  Frag16 qa[2];
  {
    const int qrow = q_base + wave * 16 + ln;
    size_t base = ((size_t)(b * 2048 + qrow)) * 3072 + (size_t)h * 64;
#pragma unroll
    for (int c = 0; c < 2; ++c) {
      qa[c].q[0] = *(const uint4*)(qkv + base + c * 32 + kb);
      qa[c].q[1] = *(const uint4*)(qkv + base + c * 32 + kb + 16);
    }
  }

  float m_i[8], l_i[8];
#pragma unroll
  for (int r = 0; r < 8; ++r) { m_i[r] = NEG_INF; l_i[r] = 0.f; }
  const v8f vzero = {0.f, 0.f, 0.f, 0.f, 0.f, 0.f, 0.f, 0.f};
  v8f o[4];
#pragma unroll
  for (int d = 0; d < 4; ++d) o[d] = vzero;

  const int lrow = tid >> 1;          // key row loaded by this thread (V tile)
  const int lseg = (tid & 1) * 32;    // dk segment

  for (int t = 0; t <= qt; ++t) {     // causal: only tiles up to the diagonal
    const int j0 = t * 64;
    __syncthreads();                  // previous tile fully consumed

    // K tile -> Ks[key][dk] via TDM; pad 128B rows by 16B (interval=4, amount=3)
    if (wave == 0) {
      const __bf16* kt = qkv + ((size_t)(b * 2048 + j0)) * 3072 + (size_t)(16 + h) * 64;
      tdm_load_2d(lds_off(Ks), kt, 64, 64, 3072, true, 4, 3);
    }
    {  // V tile -> Vt[dk][key] (manual transposed scatter, all threads)
      size_t gb = ((size_t)(b * 2048 + j0 + lrow)) * 3072 + (size_t)(32 + h) * 64 + lseg;
      union { uint4 q[4]; __bf16 hh[32]; } vv;
      const uint4* src = (const uint4*)(qkv + gb);
      vv.q[0] = src[0]; vv.q[1] = src[1]; vv.q[2] = src[2]; vv.q[3] = src[3];
#pragma unroll
      for (int d = 0; d < 32; ++d) Vt[(lseg + d) * 72 + lrow] = vv.hh[d];
    }
    if (wave == 0) __builtin_amdgcn_s_wait_tensorcnt(0);
    __syncthreads();

    // ---- scores S = Q K^T : 16 x 64 per wave (4 col-tiles x 2 k-chunks)
    v8f s[4];
#pragma unroll
    for (int nt = 0; nt < 4; ++nt) s[nt] = vzero;
#pragma unroll
    for (int kc = 0; kc < 2; ++kc) {
#pragma unroll
      for (int nt = 0; nt < 4; ++nt) {
        Frag16 kf;
        const __bf16* p = Ks + (nt * 16 + ln) * 72 + kc * 32 + kb;
        kf.q[0] = *(const uint4*)(p);
        kf.q[1] = *(const uint4*)(p + 16);
        s[nt] = wmma_bf16(qa[kc].v, kf.v, s[nt]);
      }
    }

    // ---- scale, causal mask, row max (C layout: row = 8*kg + r, col = ln)
    float rmax[8];
#pragma unroll
    for (int r = 0; r < 8; ++r) rmax[r] = NEG_INF;
#pragma unroll
    for (int nt = 0; nt < 4; ++nt) {
      const int key = j0 + nt * 16 + ln;
#pragma unroll
      for (int r = 0; r < 8; ++r) {
        const int qg = q_base + wave * 16 + 8 * kg + r;
        float v = s[nt][r] * 0.125f;  // 1/sqrt(64)
        if (key > qg) v = NEG_INF;
        s[nt][r] = v;
        rmax[r] = fmaxf(rmax[r], v);
      }
    }
#pragma unroll
    for (int msk = 1; msk < 16; msk <<= 1)
#pragma unroll
      for (int r = 0; r < 8; ++r)
        rmax[r] = fmaxf(rmax[r], __shfl_xor(rmax[r], msk, 32));

    float alpha[8], rsum[8];
#pragma unroll
    for (int r = 0; r < 8; ++r) {
      float mn = fmaxf(m_i[r], rmax[r]);
      alpha[r] = __expf(m_i[r] - mn);  // exp(-inf)=0 on first tile
      m_i[r] = mn;
      rsum[r] = 0.f;
    }
    // ---- P = exp(s - m); stage to LDS row-major for A-layout re-read
#pragma unroll
    for (int nt = 0; nt < 4; ++nt) {
#pragma unroll
      for (int r = 0; r < 8; ++r) {
        float p = __expf(s[nt][r] - m_i[r]);
        rsum[r] += p;
        Pw[wave][(8 * kg + r) * 72 + nt * 16 + ln] = (__bf16)p;
      }
    }
#pragma unroll
    for (int msk = 1; msk < 16; msk <<= 1)
#pragma unroll
      for (int r = 0; r < 8; ++r)
        rsum[r] += __shfl_xor(rsum[r], msk, 32);
#pragma unroll
    for (int r = 0; r < 8; ++r) l_i[r] = l_i[r] * alpha[r] + rsum[r];
#pragma unroll
    for (int d = 0; d < 4; ++d)
#pragma unroll
      for (int r = 0; r < 8; ++r) o[d][r] *= alpha[r];

    // P writes are wave-private; drain DS before the A-layout reads
    asm volatile("s_wait_dscnt 0" ::: "memory");

    // ---- O += P V : 16 x 64, key dim is the WMMA K dim
#pragma unroll
    for (int kc = 0; kc < 2; ++kc) {
      Frag16 pa;
      const __bf16* pp = Pw[wave] + ln * 72 + kc * 32 + kb;
      pa.q[0] = *(const uint4*)(pp);
      pa.q[1] = *(const uint4*)(pp + 16);
#pragma unroll
      for (int d = 0; d < 4; ++d) {
        Frag16 vf;
        const __bf16* vp = Vt + (d * 16 + ln) * 72 + kc * 32 + kb;
        vf.q[0] = *(const uint4*)(vp);
        vf.q[1] = *(const uint4*)(vp + 16);
        o[d] = wmma_bf16(pa.v, vf.v, o[d]);
      }
    }
  }

  // ---- normalize and store: attn[b, q, h*64 + dk] (head-concat layout)
#pragma unroll
  for (int r = 0; r < 8; ++r) {
    const int qg = q_base + wave * 16 + 8 * kg + r;
    const float inv = 1.f / l_i[r];
    size_t ob = ((size_t)(b * 2048 + qg)) * 1024 + (size_t)h * 64;
#pragma unroll
    for (int d = 0; d < 4; ++d)
      attn[ob + d * 16 + ln] = (__bf16)(o[d][r] * inv);
  }
}

// ---------------------------------------------------------------------------
// Launch: cast -> QKV GEMM -> flash attention -> output GEMM
// ---------------------------------------------------------------------------
extern "C" void kernel_launch(void* const* d_in, const int* in_sizes, int n_in,
                              void* d_out, int out_size, void* d_ws, size_t ws_size,
                              hipStream_t stream) {
  (void)in_sizes; (void)n_in; (void)out_size; (void)ws_size;
  const float* x     = (const float*)d_in[0];  // [4,2048,1024]
  const float* w_qkv = (const float*)d_in[1];  // [3,16,64,1024] = [3072,1024]
  const float* w_o   = (const float*)d_in[2];  // [1024,1024]
  float* out = (float*)d_out;                  // [4,2048,1024]

  char* ws = (char*)d_ws;
  size_t off = 0;
  auto carve = [&](size_t bytes) {
    void* p = ws + off;
    off += (bytes + 255) & ~(size_t)255;
    return p;
  };
  __bf16* xb   = (__bf16*)carve((size_t)8192 * 1024 * 2);
  __bf16* wqb  = (__bf16*)carve((size_t)3072 * 1024 * 2);
  __bf16* wob  = (__bf16*)carve((size_t)1024 * 1024 * 2);
  __bf16* qkv  = (__bf16*)carve((size_t)8192 * 3072 * 2);
  __bf16* attn = (__bf16*)carve((size_t)8192 * 1024 * 2);

  const int nX = 8192 * 1024, nWq = 3072 * 1024, nWo = 1024 * 1024;
  cast_f32_bf16<<<nX / 256, 256, 0, stream>>>(x, xb, nX);
  cast_f32_bf16<<<nWq / 256, 256, 0, stream>>>(w_qkv, wqb, nWq);
  cast_f32_bf16<<<nWo / 256, 256, 0, stream>>>(w_o, wob, nWo);

  // qkv[8192,3072] = xb[8192,1024] @ wqb[3072,1024]^T  (bf16 out)
  gemm_bf16_kernel<true><<<dim3(3072 / 128, 8192 / 64), 128, 0, stream>>>(
      xb, wqb, qkv, 8192, 3072, 1024);

  // flash attention: grid (qtiles=32, heads=16, batch=4)
  flash_attn_kernel<<<dim3(2048 / 64, 16, 4), 128, 0, stream>>>(qkv, attn);

  // out[8192,1024] = attn[8192,1024] @ wob[1024,1024]^T  (fp32 out)
  gemm_bf16_kernel<false><<<dim3(1024 / 128, 8192 / 64), 128, 0, stream>>>(
      attn, wob, out, 8192, 1024, 1024);
}
